// Amk2d_19155554140707
// MI455X (gfx1250) — matrix operand order repaired and checked by reading, not compiled, not run on hardware
//
#include <hip/hip_runtime.h>

// CDNA5 / gfx1250: wave32, WMMA 16x16x32 f16 -> f32
typedef __attribute__((ext_vector_type(16))) _Float16 v16h;
typedef __attribute__((ext_vector_type(2)))  _Float16 h2;
typedef __attribute__((ext_vector_type(8)))  float    v8f;

#define M_PTS 15      // 2^4 - 1 design points
#define P_HW  4096    // H*W = 64*64

// v_cvt_pk_f16_f32 returning our h2 type (builtin returns __fp16x2 on this
// toolchain; identical bits, so bit_cast).
static __device__ __forceinline__ h2 cvt_pk_h2(float a, float b) {
    return __builtin_bit_cast(h2, __builtin_amdgcn_cvt_pkrtz(a, b));
}

// Each wave processes a "tile pair": 32 consecutive pixels p within one (b,c)
// slice. Two 16x16x32 WMMAs compute D = chol_inv^T (16x32, padded) x k^T
// (32x16). D rows = output basis index j, D cols = pixel -> coalesced stores.
//
// The Laplace expansion exp(-|x - p_j|) = min(e^x e^{-p_j}, e^{-x} e^{p_j})
// runs in packed f16 with pairs along the j axis, so each v_pk_min_num_f16
// result IS a B-operand VGPR ({k_{2v}, k_{2v+1}}) -- no repacking moves.
__global__ __launch_bounds__(256) void laplace_wmma_kernel(
    const float* __restrict__ x,        // [B*C*P] flat (B,C,H,W)
    const float* __restrict__ pts,      // [15]
    const float* __restrict__ cholinv,  // [15,15] row-major (i = basis, j = out)
    float* __restrict__ out,            // [B, C*15, P]
    int nPairs)
{
    const int lane = threadIdx.x & 31;
    const int col  = lane & 15;   // N index (pixel within tile) / A row M
    const int hi   = lane >> 4;   // 0: A K=0..15 | 1: A K=16..31
    const float gate = hi ? 0.0f : 1.0f;  // zero B rows K=16..31 (lanes 16-31)

    // ---- Constant A = chol_inv^T, 16x32 f16, zero padded (row 15, K>=15) ----
    v16h a;
#pragma unroll
    for (int h = 0; h < 16; ++h) a[h] = (_Float16)0.0f;
#pragma unroll
    for (int h = 0; h < 16; ++h) {
        const int i = h + 16 * hi;           // basis index (K)
        float v = 0.0f;
        if (i < M_PTS && col < M_PTS) v = cholinv[i * M_PTS + col];
        a[h] = (_Float16)v;
    }

    // ---- Packed coefficient tables along j: {j=2v, j=2v+1} pairs ----
    // Lane gate folded in; j==15 slot is zero, so B element 15 and all of
    // lanes 16-31's B rows come out zero with no extra instructions.
    h2 cmh[8], cph[8];
#pragma unroll
    for (int v = 0; v < 8; ++v) {
        const int jb = 2 * v + 1;
        const float pa  = pts[2 * v];
        const float cma = __expf(-pa) * gate;
        const float cpa = __expf(pa)  * gate;
        float cmb = 0.0f, cpb = 0.0f;
        if (jb < M_PTS) {
            const float pb = pts[jb];
            cmb = __expf(-pb) * gate;
            cpb = __expf(pb)  * gate;
        }
        cmh[v] = cvt_pk_h2(cma, cmb);  // v_cvt_pk_f16_f32
        cph[v] = cvt_pk_h2(cpa, cpb);
    }

    const int wavesPerBlock = blockDim.x >> 5;
    const int w  = blockIdx.x * wavesPerBlock + (threadIdx.x >> 5);
    const int nW = gridDim.x * wavesPerBlock;
    const int pairsPerBC = P_HW / 32;

    for (int t = w; t < nPairs; t += nW) {
        const int  bc     = t / pairsPerBC;
        const int  p0     = (t - bc * pairsPerBC) * 32;
        const long inBase = (long)t * 32;    // == bc*P_HW + p0 (flat elements)

        // Prefetch next grid-stride iteration's input tile (global_prefetch_b8)
        if (t + nW < nPairs)
            __builtin_prefetch(x + (long)(t + nW) * 32 + lane, 0, 1);

        // Lane owns pixel `col` (tile 0) and pixel `col+16` (tile 1).
        const float x0 = x[inBase + col];
        const float x1 = x[inBase + 16 + col];

        // f32 exponentials, splatted to packed f16 in one v_cvt_pk_f16_f32.
        const float e0 = __expf(x0);
        const float n0 = __expf(-x0);
        const float e1 = __expf(x1);
        const float n1 = __expf(-x1);
        const h2 ex0 = cvt_pk_h2(e0, e0);
        const h2 em0 = cvt_pk_h2(n0, n0);
        const h2 ex1 = cvt_pk_h2(e1, e1);
        const h2 em1 = cvt_pk_h2(n1, n1);

        // ---- B matrices: 32x16 f16; VGPR v holds {k_{2v}, k_{2v+1}} ----
        v16h b0, b1;
#pragma unroll
        for (int v = 0; v < 8; ++v) {
            // v_pk_mul_f16 x2 + v_pk_min_num_f16, result lands in B layout.
            const h2 k0 = __builtin_elementwise_min(ex0 * cmh[v], em0 * cph[v]);
            const h2 k1 = __builtin_elementwise_min(ex1 * cmh[v], em1 * cph[v]);
            b0[2 * v]     = k0[0];
            b0[2 * v + 1] = k0[1];
            b1[2 * v]     = k1[0];
            b1[2 * v + 1] = k1[1];
        }

        // ---- D = A x B (+0):  v_wmma_f32_16x16x32_f16 ----
        v8f c0 = {};
        v8f c1 = {};
        c0 = __builtin_amdgcn_wmma_f32_16x16x32_f16(false, a, false, b0,
                                                    (short)0, c0, false, false);
        c1 = __builtin_amdgcn_wmma_f32_16x16x32_f16(false, a, false, b1,
                                                    (short)0, c1, false, false);

        // ---- Stores: out[b, c*15 + j, p] ; D reg r -> j = r + 8*hi,
        //      lanes 0-15 / 16-31 each write 64B contiguous runs.
        const long outBase = (long)bc * (M_PTS * P_HW) + p0 + col;
#pragma unroll
        for (int r = 0; r < 8; ++r) {
            const int j = r + 8 * hi;
            if (j < M_PTS) {  // drop padded row j==15 (r==7, upper half-wave)
                out[outBase + (long)j * P_HW]      = c0[r];
                out[outBase + (long)j * P_HW + 16] = c1[r];
            }
        }
    }
}

extern "C" void kernel_launch(void* const* d_in, const int* in_sizes, int n_in,
                              void* d_out, int out_size, void* d_ws, size_t ws_size,
                              hipStream_t stream) {
    const float* x       = (const float*)d_in[0];  // [16,64,64,64] fp32
    const float* pts     = (const float*)d_in[1];  // [15] fp32
    const float* cholinv = (const float*)d_in[2];  // [15,15] fp32
    float*       out     = (float*)d_out;          // [16, 64*15, 64, 64] fp32

    const int total  = in_sizes[0];   // B*C*H*W = 4,194,304
    const int nPairs = total / 32;    // 131,072 tile pairs

    const int threads = 256;          // 8 waves/block on wave32
    const int wavesPerBlock = threads / 32;
    int blocks = 1024;                // 8192 waves -> 16 pairs per wave
    const int maxBlocks = (nPairs + wavesPerBlock - 1) / wavesPerBlock;
    if (blocks > maxBlocks) blocks = maxBlocks;

    laplace_wmma_kernel<<<blocks, threads, 0, stream>>>(x, pts, cholinv, out, nPairs);
}
